// Generative_model_68762426408970
// MI455X (gfx1250) — matrix operand order repaired and checked by reading, not compiled, not run on hardware
//
#include <hip/hip_runtime.h>

// ---------- types ----------
typedef __attribute__((ext_vector_type(16))) __bf16 v16bf;
typedef __attribute__((ext_vector_type(8)))  float  v8f;
typedef __attribute__((ext_vector_type(8)))  unsigned int v8u;
typedef __attribute__((ext_vector_type(4)))  unsigned int u32x4;
typedef __attribute__((ext_vector_type(8)))  int i32x8;
typedef __attribute__((ext_vector_type(4)))  int i32x4;

union BFPack { v8u u; v16bf b; };

// ---------- constants ----------
#define VOCAB  50000
#define NT     200
#define TPAD   224      // K padded to 7*32 for GEMM2
#define MPAD1  208      // topic rows padded to 13*16 for GEMM1
#define EMB    512
#define BATCH  2048
#define TOPK   20
#define NCHUNK 25       // N-tiles per block in GEMM2
#define ROWDW  10       // padded LDS row stride in dwords (32B data + 8B pad)

// ---------- helpers ----------
__device__ __forceinline__ unsigned short f2bf(float f) {
  unsigned int x = __builtin_bit_cast(unsigned int, f);
  x = x + 0x7FFFu + ((x >> 16) & 1u);     // round-to-nearest-even
  return (unsigned short)(x >> 16);
}
__device__ __forceinline__ unsigned int pack2(float lo, float hi) {
  return (unsigned int)f2bf(lo) | ((unsigned int)f2bf(hi) << 16);
}

// ---------- kernel 1: zero Re + STDR ----------
__global__ void k_init(float* __restrict__ out) {
  int i = blockIdx.x * 256 + threadIdx.x;
  if (i < BATCH + 1) out[i] = 0.0f;
}

// ---------- kernel 2: theta = softmax(alpha) ----------
__global__ void k_theta(const float* __restrict__ alpha,
                        float* __restrict__ theta_out,
                        unsigned short* __restrict__ theta_bf) {
  const int b = blockIdx.x;
  const int t = threadIdx.x;                    // 256 threads
  __shared__ float red[256];
  float v = (t < NT) ? alpha[(size_t)b * NT + t] : -3.4e38f;
  red[t] = v; __syncthreads();
  for (int s = 128; s > 0; s >>= 1) { if (t < s) red[t] = fmaxf(red[t], red[t + s]); __syncthreads(); }
  float m = red[0]; __syncthreads();
  float e = (t < NT) ? __expf(v - m) : 0.0f;
  red[t] = e; __syncthreads();
  for (int s = 128; s > 0; s >>= 1) { if (t < s) red[t] += red[t + s]; __syncthreads(); }
  float th = e * (1.0f / red[0]);
  if (t < NT)   theta_out[(size_t)b * NT + t] = th;
  if (t < TPAD) theta_bf[(size_t)b * TPAD + t] = (t < NT) ? f2bf(th) : (unsigned short)0;
}

// ---------- kernel 3: topic embeddings -> bf16, padded rows zeroed ----------
__global__ void k_conv_topic(const float* __restrict__ te, unsigned short* __restrict__ tb) {
  int i = blockIdx.x * 256 + threadIdx.x;
  if (i < MPAD1 * EMB) {
    int r = i >> 9; // /512
    tb[i] = (r < NT) ? f2bf(te[i]) : (unsigned short)0;
  }
}

// ---------- kernel 4: GEMM1  wt[200x50000] = TE[200x512] @ WE[512x50000] ----------
// one wave (32 lanes) per 16x16 output tile, K=512 in 16 wmma steps
__global__ void __launch_bounds__(32)
k_gemm1(const unsigned short* __restrict__ topic_bf,
        const float* __restrict__ word_emb,
        float* __restrict__ wt) {
  const int l  = threadIdx.x;
  const int m0 = blockIdx.y * 16;
  const int n0 = blockIdx.x * 16;
  const int M  = l & 15;
  const int kh = l >> 4;
  const unsigned short* Arow = topic_bf + (size_t)(m0 + M) * EMB;

  v8f acc = {};
  for (int k0 = 0; k0 < EMB; k0 += 32) {
    BFPack a, bb;
#pragma unroll
    for (int v = 0; v < 8; ++v) {
      int koff = k0 + (v >> 2) * 16 + kh * 8 + (v & 3) * 2;
      a.u[v] = *(const unsigned int*)(Arow + koff);
    }
    // B: lane l = column (n0+l%16), 16 K values in table order, convert to bf16 pairs
#pragma unroll
    for (int v = 0; v < 8; ++v) {
      int kb = k0 + (v >> 2) * 16 + kh * 8 + (v & 3) * 2;
      bb.u[v] = pack2(word_emb[(size_t)kb * VOCAB + n0 + M], word_emb[(size_t)(kb + 1) * VOCAB + n0 + M]);
    }
    acc = __builtin_amdgcn_wmma_f32_16x16x32_bf16(false, a.b, false, bb.b,
                                                  (short)0, acc, false, false);
  }
  const int n = n0 + (l & 15);
#pragma unroll
  for (int r = 0; r < 8; ++r) {
    int m = m0 + (l >> 4) * 8 + r;
    if (m < NT) wt[(size_t)m * VOCAB + n] = acc[r];
  }
}

// ---------- kernel 5: beta = row-softmax(wt) in place; bf16 copy padded ----------
__global__ void k_beta(float* __restrict__ wt, unsigned short* __restrict__ beta_bf) {
  const int t   = blockIdx.x;   // 0..223
  const int tid = threadIdx.x;  // 256
  __shared__ float red[256];
  unsigned short* brow = beta_bf + (size_t)t * VOCAB;
  if (t >= NT) {
    for (int v = tid; v < VOCAB; v += 256) brow[v] = 0;
    return;
  }
  float* row = wt + (size_t)t * VOCAB;
  float m = -3.4e38f;
  for (int v = tid; v < VOCAB; v += 256) m = fmaxf(m, row[v]);
  red[tid] = m; __syncthreads();
  for (int s = 128; s > 0; s >>= 1) { if (tid < s) red[tid] = fmaxf(red[tid], red[tid + s]); __syncthreads(); }
  m = red[0]; __syncthreads();
  float sum = 0.0f;
  for (int v = tid; v < VOCAB; v += 256) sum += __expf(row[v] - m);
  red[tid] = sum; __syncthreads();
  for (int s = 128; s > 0; s >>= 1) { if (tid < s) red[tid] += red[tid + s]; __syncthreads(); }
  float inv = 1.0f / red[0];
  for (int v = tid; v < VOCAB; v += 256) {
    float bv = __expf(row[v] - m) * inv;
    row[v]  = bv;
    brow[v] = f2bf(bv);
  }
}

// ---------- kernel 6: GEMM2 fused, TDM-staged B tiles ----------
// block = 8 waves covering 128 batch rows; beta super-tile (224x16 bf16) DMA'd
// into LDS by the Tensor Data Mover, double-buffered, bank-conflict-free pad.
// Re[b] -= sum_v log((theta@beta)[b,v]) * bow[b,v]
__global__ void __launch_bounds__(256)
k_gemm2_re(const unsigned short* __restrict__ theta_bf,
           const unsigned short* __restrict__ beta_bf,
           const float* __restrict__ doc_bow,
           float* __restrict__ re_out) {
  __shared__ unsigned int sbuf[2][TPAD * ROWDW];   // 2 x 8960 B
  const int tid  = threadIdx.x;
  const int l    = tid & 31;
  const int wave = tid >> 5;                       // 0..7
  const int m0   = blockIdx.y * 128 + wave * 16;
  const int M    = l & 15;
  const int kh   = l >> 4;
  const int nt0  = blockIdx.x * NCHUNK;

  // preload this wave's A fragments for all 7 K-steps (56 VGPRs)
  unsigned int areg[7][8];
  {
    const unsigned short* Arow = theta_bf + (size_t)(m0 + M) * TPAD;
#pragma unroll
    for (int ks = 0; ks < 7; ++ks)
#pragma unroll
      for (int v = 0; v < 8; ++v) {
        int koff = ks * 32 + (v >> 2) * 16 + kh * 8 + (v & 3) * 2;
        areg[ks][v] = *(const unsigned int*)(Arow + koff);
      }
  }

  // ----- TDM descriptor (uniform) -----
  const unsigned long long gbase = (unsigned long long)(const void*)beta_bf;
  const unsigned lds0 = (unsigned)(unsigned long long)(void*)&sbuf[0][0];
  const unsigned lds1 = (unsigned)(unsigned long long)(void*)&sbuf[1][0];

  i32x8 g1;
  g1[0] = (int)((1u << 16) | (1u << 20) | (2u << 22) | (1u << 25)); // 2B elems, pad on, 32B interval, 8B pad
  g1[1] = (int)((VOCAB & 0xFFFF) << 16);                   // tensor_dim0[15:0]
  g1[2] = (int)(((VOCAB >> 16) & 0xFFFF) | (TPAD << 16));  // dim0[31:16] | dim1[15:0]
  g1[3] = (int)(16u << 16);                                // dim1[31:16]=0 | tile_dim0=16
  g1[4] = TPAD;                                            // tile_dim1 = 224
  g1[5] = VOCAB;                                           // tensor_dim0_stride[31:0]
  g1[6] = 0;
  g1[7] = 0;
  const i32x4 gz4 = {0, 0, 0, 0};
  const i32x8 gz8 = {0, 0, 0, 0, 0, 0, 0, 0};

  auto issue = [&](int nt, unsigned ldsaddr) {
    unsigned long long ga = gbase + (unsigned long long)((size_t)(nt0 + nt) * 32u); // 16 cols * 2B
    u32x4 g0;
    g0[0] = 1u;                                         // count=1, no gather
    g0[1] = ldsaddr;                                    // LDS byte address
    g0[2] = (unsigned)(ga & 0xFFFFFFFFu);
    g0[3] = (unsigned)((ga >> 32) & 0x1FFFFFFu) | (2u << 30);  // addr[56:32] | type=2
    __builtin_amdgcn_tensor_load_to_lds(g0, g1, gz4, gz4, gz8, 0);
  };

  if (tid < 32) issue(0, lds0);

  float racc[8] = {0, 0, 0, 0, 0, 0, 0, 0};

  for (int nt = 0; nt < NCHUNK; ++nt) {
    if (tid < 32) {
      if (nt + 1 < NCHUNK) {
        issue(nt + 1, ((nt + 1) & 1) ? lds1 : lds0);
        __builtin_amdgcn_s_wait_tensorcnt(1);   // tile nt complete, nt+1 in flight
      } else {
        __builtin_amdgcn_s_wait_tensorcnt(0);
      }
    }
    __syncthreads();                            // buf[nt&1] ready for all waves

    const unsigned int* bbase = &sbuf[nt & 1][0];
    v8f acc = {};
#pragma unroll
    for (int ks = 0; ks < 7; ++ks) {
      BFPack a, bb;
#pragma unroll
      for (int v = 0; v < 8; ++v) a.u[v] = areg[ks][v];
      const unsigned short* bs = (const unsigned short*)bbase + M;
#pragma unroll
      for (int v = 0; v < 8; ++v) {
        int kb = ks * 32 + (v >> 2) * 16 + kh * 8 + (v & 3) * 2;
        bb.u[v] = (unsigned int)bs[kb * (2 * ROWDW)] | ((unsigned int)bs[(kb + 1) * (2 * ROWDW)] << 16);
      }
      acc = __builtin_amdgcn_wmma_f32_16x16x32_bf16(false, a.b, false, bb.b,
                                                    (short)0, acc, false, false);
    }
    const int n = (nt0 + nt) * 16 + (l & 15);
#pragma unroll
    for (int r = 0; r < 8; ++r) {
      int m = m0 + (l >> 4) * 8 + r;
      float lg = __logf(fmaxf(acc[r], 1e-38f));
      racc[r] += lg * doc_bow[(size_t)m * VOCAB + n];
    }
    __syncthreads();                            // all done reading buf[nt&1]
  }

  // reduce over the 16 lanes of each half (same M), then atomic into -Re
#pragma unroll
  for (int r = 0; r < 8; ++r) {
    float s = racc[r];
    s += __shfl_xor(s, 1);
    s += __shfl_xor(s, 2);
    s += __shfl_xor(s, 4);
    s += __shfl_xor(s, 8);
    if ((l & 15) == 0) {
      int m = m0 + (l >> 4) * 8 + r;
      atomicAdd(&re_out[m], -s);
    }
  }
}

// ---------- kernel 7: per-row top-20 of beta (mutates ws beta copy) ----------
__global__ void k_topk(float* __restrict__ beta,
                       float* __restrict__ tk_vals, int* __restrict__ tk_idx) {
  const int t   = blockIdx.x;   // 200
  const int tid = threadIdx.x;  // 256
  __shared__ float rv[256];
  __shared__ int   ri[256];
  float* row = beta + (size_t)t * VOCAB;
  for (int j = 0; j < TOPK; ++j) {
    float bm = -2.0f; int bi = VOCAB;
    for (int v = tid; v < VOCAB; v += 256) {
      float x = row[v];
      if (x > bm || (x == bm && v < bi)) { bm = x; bi = v; }
    }
    rv[tid] = bm; ri[tid] = bi; __syncthreads();
    for (int s = 128; s > 0; s >>= 1) {
      if (tid < s) {
        if (rv[tid + s] > rv[tid] || (rv[tid + s] == rv[tid] && ri[tid + s] < ri[tid])) {
          rv[tid] = rv[tid + s]; ri[tid] = ri[tid + s];
        }
      }
      __syncthreads();
    }
    if (tid == 0) {
      tk_vals[t * TOPK + j] = rv[0];
      tk_idx [t * TOPK + j] = ri[0];
      row[ri[0]] = -1.0f;          // poison; beta is recomputed every launch
    }
    __syncthreads();
  }
  if (tid == 0) {
    float s = 0.0f;
    for (int j = 0; j < TOPK; ++j) s += tk_vals[t * TOPK + j];
    float inv = 1.0f / s;
    for (int j = 0; j < TOPK; ++j) tk_vals[t * TOPK + j] *= inv;
  }
}

// ---------- kernel 8: STDR via sparse Gram ----------
__global__ void k_stdr(const float* __restrict__ tk_vals, const int* __restrict__ tk_idx,
                       float* __restrict__ stdr_out) {
  const int i = blockIdx.x;     // 200
  const int j = threadIdx.x;    // 256
  __shared__ float vi[TOPK];
  __shared__ int   ii[TOPK];
  __shared__ float red[256];
  if (j < TOPK) { vi[j] = tk_vals[i * TOPK + j]; ii[j] = tk_idx[i * TOPK + j]; }
  __syncthreads();
  float c = 0.0f;
  if (j < NT) {
    float sqi = 0.0f, sqj = 0.0f, dot = 0.0f;
    float vjv[TOPK]; int iji[TOPK];
#pragma unroll
    for (int a = 0; a < TOPK; ++a) {
      vjv[a] = tk_vals[j * TOPK + a];
      iji[a] = tk_idx [j * TOPK + a];
      sqj += vjv[a] * vjv[a];
      sqi += vi[a] * vi[a];
    }
    for (int a = 0; a < TOPK; ++a) {
      int ia = ii[a]; float va = vi[a];
#pragma unroll
      for (int b = 0; b < TOPK; ++b)
        if (iji[b] == ia) dot += va * vjv[b];
    }
    float d2 = fmaxf(sqi + sqj - 2.0f * dot, 0.0f);
    c = 0.5f * d2;
  }
  red[j] = c; __syncthreads();
  for (int s = 128; s > 0; s >>= 1) { if (j < s) red[j] += red[j + s]; __syncthreads(); }
  if (j == 0) atomicAdd(stdr_out, red[0] * (1.0f / ((float)NT * (float)NT)));
}

// ---------- launcher ----------
extern "C" void kernel_launch(void* const* d_in, const int* in_sizes, int n_in,
                              void* d_out, int out_size, void* d_ws, size_t ws_size,
                              hipStream_t stream) {
  const float* alpha     = (const float*)d_in[0];   // [2048,200]
  const float* doc_bow   = (const float*)d_in[1];   // [2048,50000]
  const float* topic_emb = (const float*)d_in[2];   // [200,512]
  const float* word_emb  = (const float*)d_in[3];   // [512,50000]

  float* out = (float*)d_out;       // [Re 2048 | STDR 1 | theta 2048*200]
  float* re_out    = out;
  float* stdr_out  = out + BATCH;
  float* theta_out = out + BATCH + 1;

  // workspace layout (128B aligned chunks)
  char* ws = (char*)d_ws;
  unsigned short* theta_bf = (unsigned short*)(ws);                         // 2048*224*2   = 917504
  unsigned short* topic_bf = (unsigned short*)(ws + 917504);                // 208*512*2    = 212992
  float*          wt       = (float*)(ws + 1130496);                        // 200*50000*4  = 40000000 (becomes beta)
  unsigned short* beta_bf  = (unsigned short*)(ws + 41130496);              // 224*50000*2  = 22400000
  float*          tk_vals  = (float*)(ws + 63530496);                       // 200*20*4
  int*            tk_idx   = (int*)  (ws + 63546496);                       // 200*20*4

  k_init<<<(BATCH + 1 + 255) / 256, 256, 0, stream>>>(out);
  k_theta<<<BATCH, 256, 0, stream>>>(alpha, theta_out, theta_bf);
  k_conv_topic<<<(MPAD1 * EMB + 255) / 256, 256, 0, stream>>>(topic_emb, topic_bf);
  k_gemm1<<<dim3(VOCAB / 16, MPAD1 / 16), 32, 0, stream>>>(topic_bf, word_emb, wt);
  k_beta<<<TPAD, 256, 0, stream>>>(wt, beta_bf);
  k_gemm2_re<<<dim3(125, BATCH / 128), 256, 0, stream>>>(theta_bf, beta_bf, doc_bow, re_out);
  k_topk<<<NT, 256, 0, stream>>>(wt, tk_vals, tk_idx);
  k_stdr<<<NT, 256, 0, stream>>>(tk_vals, tk_idx, stdr_out);
}
